// BiLSTM_CRF_32160715112573
// MI455X (gfx1250) — matrix operand order repaired and checked by reading, compiled, not run on hardware
//
#include <hip/hip_runtime.h>
#include <hip/hip_bf16.h>
#include <math.h>

// ---------------- problem constants ----------------
#define B_      8
#define S_      2048
#define E_      256
#define HH_     256
#define G4_     1024          // 4*HH
#define T_      12
#define NROWS_  16384         // S*B
#define START_TAG 10
#define STOP_TAG  11
#define NEGV     (-10000.0f)

typedef __bf16 bf16;
typedef __attribute__((ext_vector_type(16))) __bf16 v16bf;
typedef __attribute__((ext_vector_type(8)))  __bf16 v8bf;
typedef __attribute__((ext_vector_type(8)))  float  v8f;

// ---------------- workspace layout (bytes, 256B aligned) ----------------
#define WS_XS     ((size_t)0)            // bf16 [16384][256]   embedded inputs, row r=s*8+b
#define WS_WIH    ((size_t)8388608)      // bf16 [2][1024][256] input weights
#define WS_WHH    ((size_t)9437184)      // bf16 [2][1024][256] recurrent weights
#define WS_XPROJ  ((size_t)10485760)     // bf16 [2][16384][1024]  x@Wih.T + b
#define WS_HOUT   ((size_t)77594624)     // f32  [2][16384][256]   hidden outputs (r=s*8+b)
#define WS_FEATS  ((size_t)111149056)    // f32  [16384][12]       tag scores (row r'=b*S+s)
#define WS_BPTR   ((size_t)111935488)    // u8   [16384][12]       viterbi backpointers

// ---------------- WMMA fragment loaders (bf16 16x16x32) ----------------
// A (16x32, MxK) from row-major [M][ld] storage.
// lanes 0-15:  M=lane,    K = {0..7, 16..23}
// lanes 16-31: M=lane-16, K = {8..15, 24..31}
__device__ inline v16bf load_frag_a(const bf16* __restrict__ base, int ld, int lane) {
  int m  = lane & 15;
  int kb = (lane >> 4) * 8;
  const bf16* p = base + (size_t)m * ld + kb;
  union { v16bf v; v8bf h[2]; } u;
  u.h[0] = *(const v8bf*)(p);
  u.h[1] = *(const v8bf*)(p + 16);
  return u.v;
}
// B (32x16, KxN) where B[k][n] = w[n][k], w row-major [N][ldw] (weight layout).
// lanes 0-15:  N=lane,    K = {0..7, 16..23} ; lanes 16-31: N=lane-16, K = {8..15, 24..31}
__device__ inline v16bf load_frag_b(const bf16* __restrict__ w, int ldw, int lane) {
  int n  = lane & 15;
  int kb = (lane >> 4) * 8;
  const bf16* p = w + (size_t)n * ldw + kb;
  union { v16bf v; v8bf h[2]; } u;
  u.h[0] = *(const v8bf*)(p);
  u.h[1] = *(const v8bf*)(p + 16);
  return u.v;
}

__device__ inline float sigm(float x) { return 1.0f / (1.0f + __expf(-x)); }

// ---------------- K0a: convert weights to bf16 ----------------
__global__ void k_cvt_weights(const float* __restrict__ wihf, const float* __restrict__ whhf,
                              const float* __restrict__ wihb, const float* __restrict__ whhb,
                              bf16* __restrict__ wih, bf16* __restrict__ whh) {
  int i = blockIdx.x * blockDim.x + threadIdx.x;
  if (i >= G4_ * E_) return;
  wih[i]             = (bf16)wihf[i];
  wih[G4_ * E_ + i]  = (bf16)wihb[i];
  whh[i]             = (bf16)whhf[i];
  whh[G4_ * E_ + i]  = (bf16)whhb[i];
}

// ---------------- K0b: embedding gather -> bf16, row r = s*8+b ----------------
__global__ void k_gather(const int* __restrict__ sent, const float* __restrict__ emb,
                         bf16* __restrict__ xs) {
  int i = blockIdx.x * blockDim.x + threadIdx.x;
  if (i >= NROWS_ * E_) return;
  int r = i >> 8, e = i & 255;
  int s = r >> 3, b = r & 7;
  int tok = sent[b * S_ + s];
  xs[i] = (bf16)emb[(size_t)tok * E_ + e];
}

// ---------------- K1: input projection GEMM (both dirs), bias fused ----------------
// xproj[dir][r][n] = bf16( sum_k xs[r][k]*wih[dir][n][k] + bias[dir][n] )
__global__ __launch_bounds__(256) void k_in_gemm(
    const bf16* __restrict__ xs, const bf16* __restrict__ wih,
    const float* __restrict__ b_f, const float* __restrict__ b_b,
    bf16* __restrict__ xproj) {
  int wave = (blockIdx.x * blockDim.x + threadIdx.x) >> 5;   // 131072 waves
  int lane = threadIdx.x & 31;
  int dir   = wave >> 16;
  int tile  = wave & 65535;
  int mtile = tile >> 6;        // 1024 M-tiles of 16 rows
  int ntile = tile & 63;        // 64 N-tiles of 16 cols

  const bf16* w = wih + (size_t)dir * (G4_ * E_) + (size_t)(ntile * 16) * E_;
  const bf16* a0 = xs + (size_t)(mtile * 16) * E_;

  v8f acc = {0.f, 0.f, 0.f, 0.f, 0.f, 0.f, 0.f, 0.f};
#pragma unroll
  for (int kt = 0; kt < 8; ++kt) {
    v16bf a = load_frag_a(a0 + kt * 32, E_, lane);
    v16bf b = load_frag_b(w + kt * 32, E_, lane);
    acc = __builtin_amdgcn_wmma_f32_16x16x32_bf16(false, a, false, b, (short)0, acc,
                                                  false, false);
  }
  const float* bias = dir ? b_b : b_f;
  int ncol  = ntile * 16 + (lane & 15);
  float bv  = bias[ncol];
  int mbase = (lane >> 4) * 8;                 // C layout: VGPR v -> M = v + 8*(lane>=16)
  bf16* out = xproj + (size_t)dir * ((size_t)NROWS_ * G4_);
#pragma unroll
  for (int v = 0; v < 8; ++v) {
    int m = mbase + v;
    out[(size_t)(mtile * 16 + m) * G4_ + ncol] = (bf16)(acc[v] + bv);
  }
}

// ---------------- K2: serial recurrent LSTM, 1 workgroup per direction ----------------
__global__ __launch_bounds__(1024) void k_lstm(
    const bf16* __restrict__ whh, const bf16* __restrict__ xproj,
    const float* __restrict__ h0, const float* __restrict__ c0,
    float* __restrict__ hout) {
  __shared__ bf16  h_sh[16][256];   // rows 8..15 stay zero (M padding for WMMA)
  __shared__ float c_sh[8][256];
  __shared__ float z_sh[8][1024];   // gate pre-activations from h@Whh.T

  const int dir  = blockIdx.x;
  const int tid  = threadIdx.x;
  const int lane = tid & 31;
  const int wv   = tid >> 5;        // 32 waves

  if (tid < 2048) {
    int b = tid >> 8, u = tid & 255;
    h_sh[b][u]     = (bf16)h0[dir * 2048 + tid];
    h_sh[8 + b][u] = (bf16)0.0f;
    c_sh[b][u]     = c0[dir * 2048 + tid];
  }
  __syncthreads();

  const bf16* w  = whh + (size_t)dir * (G4_ * E_);
  const bf16* xp = xproj + (size_t)dir * ((size_t)NROWS_ * G4_);
  float*      ho = hout + (size_t)dir * ((size_t)NROWS_ * HH_);

  for (int it = 0; it < S_; ++it) {
    int s = dir ? (S_ - 1 - it) : it;

    // z = h @ Whh.T : M=16(pad of 8), N=1024, K=256 ; wave handles 2 N-tiles
#pragma unroll
    for (int q = 0; q < 2; ++q) {
      int nt = wv * 2 + q;
      v8f acc = {0.f, 0.f, 0.f, 0.f, 0.f, 0.f, 0.f, 0.f};
#pragma unroll
      for (int kt = 0; kt < 8; ++kt) {
        v16bf a = load_frag_a(&h_sh[0][0] + kt * 32, 256, lane);
        v16bf b = load_frag_b(w + (size_t)(nt * 16) * E_ + kt * 32, E_, lane);
        acc = __builtin_amdgcn_wmma_f32_16x16x32_bf16(false, a, false, b, (short)0, acc,
                                                      false, false);
      }
      int ncol  = nt * 16 + (lane & 15);
      int mbase = (lane >> 4) * 8;
#pragma unroll
      for (int v = 0; v < 8; ++v) {
        int m = mbase + v;
        if (m < 8) z_sh[m][ncol] = acc[v];
      }
    }
    __syncthreads();

    // elementwise LSTM cell (PyTorch gate order i,f,g,o)
    if (tid < 2048) {
      int b = tid >> 8, u = tid & 255;
      size_t r = (size_t)s * 8 + b;
      const bf16* x = xp + r * G4_;
      float zi = z_sh[b][u]       + (float)x[u];
      float zf = z_sh[b][256 + u] + (float)x[256 + u];
      float zg = z_sh[b][512 + u] + (float)x[512 + u];
      float zo = z_sh[b][768 + u] + (float)x[768 + u];
      float cc = sigm(zf) * c_sh[b][u] + sigm(zi) * tanhf(zg);
      float hh = sigm(zo) * tanhf(cc);
      c_sh[b][u] = cc;
      h_sh[b][u] = (bf16)hh;
      ho[r * HH_ + u] = hh;
    }
    __syncthreads();
  }
}

// ---------------- K3: feats = [hf|hb] @ W_out.T + b_out, row order r'=b*S+s ----------------
__global__ __launch_bounds__(256) void k_feats(
    const float* __restrict__ hf, const float* __restrict__ hb,
    const float* __restrict__ W_out, const float* __restrict__ b_out,
    float* __restrict__ feats) {
  int g = blockIdx.x * blockDim.x + threadIdx.x;
  int rp = g >> 4;            // flattened chain row: b*S + s
  int t  = g & 15;
  if (rp >= NROWS_ || t >= T_) return;
  int b = rp >> 11, s = rp & 2047;
  size_t r = (size_t)s * 8 + b;           // storage row of hidden states
  const float* wr = W_out + t * 512;
  const float* pf = hf + r * 256;
  const float* pb = hb + r * 256;
  float acc = b_out[t];
  for (int u = 0; u < 256; ++u) acc += wr[u] * pf[u] + wr[256 + u] * pb[u];
  feats[(size_t)rp * T_ + t] = acc;
}

// ---------------- K4: Viterbi forward + backtrace (1 wave) ----------------
__global__ __launch_bounds__(32) void k_viterbi(
    const float* __restrict__ feats, const float* __restrict__ trans,
    unsigned char* __restrict__ bptr, float* __restrict__ out) {
  __shared__ float fv[16];
  __shared__ float tr[12][12];
  int t = threadIdx.x;
  if (t < T_) {
    fv[t] = (t == START_TAG) ? 0.0f : NEGV;
    for (int p = 0; p < T_; ++p) tr[t][p] = trans[t * T_ + p];
  }
  __syncthreads();

  for (int r = 0; r < NROWS_; ++r) {
    float nf = 0.0f;
    if (t < T_) {
      float best = -3.4e38f; int bp = 0;
      for (int p = 0; p < T_; ++p) {
        float sc = fv[p] + tr[t][p];       // scores[next=t][prev=p]
        if (sc > best) { best = sc; bp = p; }
      }
      bptr[r * T_ + t] = (unsigned char)bp;
      nf = best + feats[r * T_ + t];
    }
    __syncthreads();
    if (t < T_) fv[t] = nf;
    __syncthreads();
  }

  if (t == 0) {
    float bs = -3.4e38f; int bt = 0;
    for (int p = 0; p < T_; ++p) {
      float sc = fv[p] + tr[STOP_TAG][p];
      if (sc > bs) { bs = sc; bt = p; }
    }
    out[0] = bs;                       // path_score
    out[NROWS_] = (float)bt;           // best_path[N-1] = best tag
    int tag = bt;
    for (int r = NROWS_ - 1; r >= 1; --r) {
      int prev = bptr[r * T_ + tag];   // path[r]
      out[r] = (float)prev;            // best_path[r-1]
      tag = prev;
    }
  }
}

// ---------------- launch ----------------
extern "C" void kernel_launch(void* const* d_in, const int* in_sizes, int n_in,
                              void* d_out, int out_size, void* d_ws, size_t ws_size,
                              hipStream_t stream) {
  const int*   sentence = (const int*)d_in[0];
  const float* emb      = (const float*)d_in[1];
  const float* w_ih_f   = (const float*)d_in[2];
  const float* w_hh_f   = (const float*)d_in[3];
  const float* b_f      = (const float*)d_in[4];
  const float* w_ih_b   = (const float*)d_in[5];
  const float* w_hh_b   = (const float*)d_in[6];
  const float* b_b      = (const float*)d_in[7];
  const float* h0       = (const float*)d_in[8];
  const float* c0       = (const float*)d_in[9];
  const float* W_out    = (const float*)d_in[10];
  const float* b_out    = (const float*)d_in[11];
  const float* trans    = (const float*)d_in[12];
  float* out = (float*)d_out;

  char* ws = (char*)d_ws;
  bf16*  xs    = (bf16*)(ws + WS_XS);
  bf16*  wih   = (bf16*)(ws + WS_WIH);
  bf16*  whh   = (bf16*)(ws + WS_WHH);
  bf16*  xproj = (bf16*)(ws + WS_XPROJ);
  float* hout  = (float*)(ws + WS_HOUT);
  float* feats = (float*)(ws + WS_FEATS);
  unsigned char* bptr = (unsigned char*)(ws + WS_BPTR);

  // K0: weight conversion + embedding gather
  k_cvt_weights<<<(G4_ * E_ + 255) / 256, 256, 0, stream>>>(w_ih_f, w_hh_f, w_ih_b, w_hh_b,
                                                            wih, whh);
  k_gather<<<(NROWS_ * E_ + 255) / 256, 256, 0, stream>>>(sentence, emb, xs);

  // K1: batched input projection (both directions), 131072 waves / 8 per block
  k_in_gemm<<<16384, 256, 0, stream>>>(xs, wih, b_f, b_b, xproj);

  // K2: serial bidirectional LSTM, one workgroup per direction
  k_lstm<<<2, 1024, 0, stream>>>(whh, xproj, h0, c0, hout);

  // K3: tag-score projection
  k_feats<<<(NROWS_ * 16) / 256, 256, 0, stream>>>(hout, hout + (size_t)NROWS_ * HH_,
                                                   W_out, b_out, feats);

  // K4: Viterbi decode
  k_viterbi<<<1, 32, 0, stream>>>(feats, trans, bptr, out);
}
